// TopKRouter_79285096284329
// MI455X (gfx1250) — matrix operand order repaired and checked by reading, not compiled, not run on hardware
//
#include <hip/hip_runtime.h>
#include <hip/hip_bf16.h>
#include <math.h>

#define DIM    4096
#define NEXP   64
#define TOPK   8
#define TOKENS 16384
#define PITCH  68   // LDS row pitch in floats: 16B-aligned rows, staggered banks

typedef __attribute__((ext_vector_type(16))) __bf16 v16bf;
typedef __attribute__((ext_vector_type(4)))  __bf16 v4bf;
typedef __attribute__((ext_vector_type(8)))  float  v8f;

// ---------------------------------------------------------------------------
// W fp32 [64,4096] -> bf16 into workspace (done every launch; 1 MB read).
// ---------------------------------------------------------------------------
__global__ __launch_bounds__(256) void convert_w_kernel(const float* __restrict__ W,
                                                        __bf16* __restrict__ Wb) {
  int i = blockIdx.x * blockDim.x + threadIdx.x;   // 4 elements per thread
  float4 f = ((const float4*)W)[i];
  v4bf o;
  o[0] = (__bf16)f.x; o[1] = (__bf16)f.y; o[2] = (__bf16)f.z; o[3] = (__bf16)f.w;
  ((v4bf*)Wb)[i] = o;
}

// Insert (v,i) into a sorted-descending top-8 list. Strict '>' so that for
// equal values the earlier (lower-index) entry stays higher, matching
// jax.lax.top_k tie-breaking when values are fed in ascending index order.
__device__ __forceinline__ void insert8(float v, int i, float (&vals)[8], int (&idxs)[8]) {
#pragma unroll
  for (int s = 0; s < 8; ++s) {
    bool g = v > vals[s];
    float tv = vals[s]; int ti = idxs[s];
    vals[s] = g ? v : tv;   idxs[s] = g ? i : ti;
    v       = g ? tv : v;   i       = g ? ti : i;
  }
}

// ---------------------------------------------------------------------------
// Fused router: bf16 WMMA gate GEMM (fp32 accum) + top-8 + softmax.
// Block = 256 threads = 8 waves = 4 wave-pairs. Each pair: 16 tokens, K split
// in halves across the two waves, reduced through LDS float atomics.
// ---------------------------------------------------------------------------
__global__ __launch_bounds__(256) void router_kernel(const float* __restrict__ x,
                                                     const __bf16* __restrict__ Wb,
                                                     float* __restrict__ scores,
                                                     int* __restrict__ inds) {
  __shared__ float lds[4][16][PITCH];

  const int tid  = threadIdx.x;
  const int wave = tid >> 5;
  const int lane = tid & 31;
  const int pair = wave >> 1;   // 0..3 : 16-token tile within block
  const int kh   = wave & 1;    // K-split half
  const int n    = lane & 15;   // row (A) / column (B) index within tile
  const int half = lane >> 4;

  // Zero the logit accumulation area (LDS is uninitialized).
  float* lz = &lds[0][0][0];
  for (int i = tid; i < 4 * 16 * PITCH; i += 256) lz[i] = 0.0f;
  __syncthreads();

  const int tokBase = blockIdx.x * 64 + pair * 16;
  const float* xrow = x + (size_t)(tokBase + n) * DIM;
  const __bf16* wr0 = Wb + (size_t)(0 * 16 + n) * DIM;
  const __bf16* wr1 = Wb + (size_t)(1 * 16 + n) * DIM;
  const __bf16* wr2 = Wb + (size_t)(2 * 16 + n) * DIM;
  const __bf16* wr3 = Wb + (size_t)(3 * 16 + n) * DIM;

  v8f acc0 = {}, acc1 = {}, acc2 = {}, acc3 = {};

  const int kBeg = kh * (DIM / 2);
  const int kEnd = kBeg + (DIM / 2);
#pragma unroll 2
  for (int k0 = kBeg; k0 < kEnd; k0 += 32) {
    // A fragment (16x32 bf16): lanes<16 hold K 0..15,
    // lanes>=16 hold K 16..31 of row M = lane&15 (matches B).
    const int ka = k0 + half * 16;
    float4 f0 = *(const float4*)(xrow + ka);
    float4 f1 = *(const float4*)(xrow + ka + 4);
    float4 f2 = *(const float4*)(xrow + ka + 8);
    float4 f3 = *(const float4*)(xrow + ka + 12);
    v16bf a;
    a[0]  = (__bf16)f0.x; a[1]  = (__bf16)f0.y; a[2]  = (__bf16)f0.z; a[3]  = (__bf16)f0.w;
    a[4]  = (__bf16)f1.x; a[5]  = (__bf16)f1.y; a[6]  = (__bf16)f1.z; a[7]  = (__bf16)f1.w;
    a[8]  = (__bf16)f2.x; a[9]  = (__bf16)f2.y; a[10] = (__bf16)f2.z; a[11] = (__bf16)f2.w;
    a[12] = (__bf16)f3.x; a[13] = (__bf16)f3.y; a[14] = (__bf16)f3.z; a[15] = (__bf16)f3.w;

    // B fragments (32x16 bf16): lane = column n, 16 contiguous K per lane
    // (lanes<16: K 0..15, lanes>=16: K 16..31) -> aligned 32B loads.
    const int kb = k0 + half * 16;
    v16bf b0 = *(const v16bf*)(wr0 + kb);
    acc0 = __builtin_amdgcn_wmma_f32_16x16x32_bf16(false, a, false, b0, (short)0, acc0, false, false);
    v16bf b1 = *(const v16bf*)(wr1 + kb);
    acc1 = __builtin_amdgcn_wmma_f32_16x16x32_bf16(false, a, false, b1, (short)0, acc1, false, false);
    v16bf b2 = *(const v16bf*)(wr2 + kb);
    acc2 = __builtin_amdgcn_wmma_f32_16x16x32_bf16(false, a, false, b2, (short)0, acc2, false, false);
    v16bf b3 = *(const v16bf*)(wr3 + kb);
    acc3 = __builtin_amdgcn_wmma_f32_16x16x32_bf16(false, a, false, b3, (short)0, acc3, false, false);
  }

  // K-split reduction: C layout is VGPR j -> M = half*8 + j, N = e*16 + n.
#pragma unroll
  for (int j = 0; j < 8; ++j) {
    const int row = half * 8 + j;
    atomicAdd(&lds[pair][row][0 * 16 + n], acc0[j]);
    atomicAdd(&lds[pair][row][1 * 16 + n], acc1[j]);
    atomicAdd(&lds[pair][row][2 * 16 + n], acc2[j]);
    atomicAdd(&lds[pair][row][3 * 16 + n], acc3[j]);
  }
  __syncthreads();

  if (kh == 0) {  // one wave per pair does the top-k (wave-uniform branch)
    // Lane (n, half) scans row n over columns [half*32, half*32+32).
    float vals[8]; int idxs[8];
#pragma unroll
    for (int s = 0; s < 8; ++s) { vals[s] = -INFINITY; idxs[s] = 0; }
    const int cbase = half * 32;
    const float4* lp = (const float4*)&lds[pair][n][cbase];
#pragma unroll
    for (int c4 = 0; c4 < 8; ++c4) {
      float4 v = lp[c4];
      insert8(v.x, cbase + c4 * 4 + 0, vals, idxs);
      insert8(v.y, cbase + c4 * 4 + 1, vals, idxs);
      insert8(v.z, cbase + c4 * 4 + 2, vals, idxs);
      insert8(v.w, cbase + c4 * 4 + 3, vals, idxs);
    }
    // Merge upper-column half (lanes 16..31) into lanes 0..15. Partner
    // indices are all larger, so strict '>' keeps tie order correct.
#pragma unroll
    for (int s = 0; s < 8; ++s) {
      float pv = __shfl(vals[s], lane + 16, 32);
      int   pi = __shfl(idxs[s], lane + 16, 32);
      if (lane < 16) insert8(pv, pi, vals, idxs);
    }
    if (lane < 16) {
      const int token = tokBase + n;
      const float m = vals[0];
      float e[8]; float sum = 0.0f;
#pragma unroll
      for (int s = 0; s < 8; ++s) { e[s] = __expf(vals[s] - m); sum += e[s]; }
      const float inv = 1.0f / sum;
#pragma unroll
      for (int s = 0; s < 8; ++s) {
        scores[(size_t)token * TOPK + s] = e[s] * inv;
        inds[(size_t)token * TOPK + s]   = idxs[s];
      }
    }
  }
}

extern "C" void kernel_launch(void* const* d_in, const int* in_sizes, int n_in,
                              void* d_out, int out_size, void* d_ws, size_t ws_size,
                              hipStream_t stream) {
  const float* x = (const float*)d_in[0];
  const float* W = (const float*)d_in[1];
  __bf16* Wb = (__bf16*)d_ws;                       // 64*4096*2 = 512 KB scratch
  float* scores = (float*)d_out;                    // [16384, 8] fp32
  int*   inds   = (int*)d_out + (size_t)TOKENS * TOPK;  // [16384, 8] int32 (bit view)

  convert_w_kernel<<<(NEXP * DIM / 4) / 256, 256, 0, stream>>>(W, Wb);
  router_kernel<<<TOKENS / 64, 256, 0, stream>>>(x, Wb, scores, inds);
}
